// monkey_switch_group_QKV_three_22539988369549
// MI455X (gfx1250) — matrix operand order repaired and checked by reading, not compiled, not run on hardware
//
#include <hip/hip_runtime.h>
#include <hip/hip_bf16.h>
#include <stdint.h>

typedef __bf16 bf16;
typedef __attribute__((ext_vector_type(16))) __bf16 v16bf;
typedef __attribute__((ext_vector_type(8)))  float  v8f;

#define GN_COLS 4096   // N dim of conv GEMMs: B(64) * HW(64)
#define CH      512
#define HW_SZ   64

struct __align__(16) U128 { unsigned int w[4]; };
union FragCvt { U128 u[2]; v16bf f; };

// ---- gfx1250 async global->LDS via inline asm (ISA: GLOBAL_LOAD_ASYNC_TO_LDS)
// Per-lane 16B copy global->LDS, tracked by ASYNCcnt. Falls back to a plain
// b128 copy when not compiling for gfx1250 (host pass).
__device__ __forceinline__ void stage_copy_b128(const bf16* gsrc, bf16* ldst) {
#if defined(__gfx1250__)
  unsigned lds_off =
      (unsigned)(size_t)((__attribute__((address_space(3))) char*)(void*)ldst);
  asm volatile("global_load_async_to_lds_b128 %0, %1, off"
               :
               : "v"(lds_off), "v"(gsrc)
               : "memory");
#else
  *(U128*)ldst = *(const U128*)gsrc;
#endif
}

__device__ __forceinline__ void wait_async_lds() {
#if defined(__gfx1250__)
#if defined(__has_builtin) && __has_builtin(__builtin_amdgcn_s_wait_asynccnt)
  __builtin_amdgcn_s_wait_asynccnt(0);
#else
  asm volatile("s_wait_asynccnt 0x0" ::: "memory");
#endif
#endif
}

// A fragment (16x32 bf16, MxK), source LDS layout [row][rowlen] row-major.
// Lane L: M = row (caller passes tile_base + (L&15)), half = L>>4.
// VGPR 0..3 hold K = 8*half .. 8*half+7 ; VGPR 4..7 hold K = 16+8*half .. +7.
__device__ __forceinline__ v16bf frag_a(const bf16* base, int row, int rowlen, int k0) {
  const int half = (threadIdx.x & 31) >> 4;
  const bf16* p = base + row * rowlen + k0 + 8 * half;
  FragCvt c;
  c.u[0] = *(const U128*)p;
  c.u[1] = *(const U128*)(p + 16);
  return c.f;
}
// B fragment (32x16 bf16, KxN), source LDS layout [n][rowlen] (n-major).
// Lane L: N = n (caller passes tile_base + (L&15)), half = L>>4.
// VGPR 0..3 hold K = 8*half .. 8*half+7 ; VGPR 4..7 hold K = 16+8*half .. +7.
__device__ __forceinline__ v16bf frag_b(const bf16* base, int n, int rowlen, int k0) {
  const int half = (threadIdx.x & 31) >> 4;
  const bf16* p = base + n * rowlen + k0 + 8 * half;
  FragCvt c;
  c.u[0] = *(const U128*)p;
  c.u[1] = *(const U128*)(p + 16);
  return c.f;
}
__device__ __forceinline__ v8f wmma_bf16(v16bf a, v16bf b, v8f c) {
  return __builtin_amdgcn_wmma_f32_16x16x32_bf16(false, a, false, b, (short)0, c,
                                                 false, false);
}

// ---------------------------------------------------------------------------
// im2col: X f32 [B, Cin, 8, 8] -> Bmat bf16 [Kp, 4096], k = ci*9+kh*3+kw,
// n = b*64 + h*8 + w, SAME padding, rows k >= Cin*9 zero-filled.
// ---------------------------------------------------------------------------
__global__ void __launch_bounds__(256)
im2col_bf16(const float* __restrict__ x, bf16* __restrict__ Bm, int Cin, int Kp) {
  size_t idx = (size_t)blockIdx.x * 256 + threadIdx.x;
  if (idx >= (size_t)Kp * GN_COLS) return;
  int n = (int)(idx & (GN_COLS - 1));
  int k = (int)(idx >> 12);
  float v = 0.f;
  if (k < Cin * 9) {
    int ci = k / 9;
    int r  = k - ci * 9;
    int kh = r / 3, kw = r - kh * 3;
    int b = n >> 6, hw = n & 63;
    int h = hw >> 3, w = hw & 7;
    int ih = h + kh - 1, iw = w + kw - 1;
    if ((unsigned)ih < 8u && (unsigned)iw < 8u)
      v = x[(((size_t)b * Cin + ci) << 6) + ih * 8 + iw];
  }
  Bm[idx] = (bf16)v;
}

// Weight fp32 [512][Cin*9] -> bf16 [512][Kp], K-padded with zeros.
__global__ void __launch_bounds__(256)
weight_to_bf16(const float* __restrict__ w, bf16* __restrict__ Aw, int K, int Kp) {
  size_t idx = (size_t)blockIdx.x * 256 + threadIdx.x;
  if (idx >= (size_t)CH * Kp) return;
  int kk = (int)(idx % Kp);
  int m  = (int)(idx / Kp);
  Aw[idx] = (bf16)((kk < K) ? w[(size_t)m * K + kk] : 0.f);
}

// ---------------------------------------------------------------------------
// Implicit-GEMM conv: out(NCHW f32) = A[512,Kp](bf16) x Bmat[Kp,4096](bf16).
// 256 thr / 8 waves, block tile 128Mx64N, wave tile 32x32, K-step 32.
// A tile staged with GLOBAL_LOAD_ASYNC_TO_LDS_B128 (ASYNCcnt); next K-step
// rows pulled ahead with global_prefetch_b8 while WMMAs execute.
// ---------------------------------------------------------------------------
__global__ void __launch_bounds__(256)
gemm_conv_wmma(const bf16* __restrict__ A, const bf16* __restrict__ Bm,
               float* __restrict__ out, int M, int Kp) {
  __shared__ __align__(16) bf16 sA[128 * 32];   // [m][k]
  __shared__ __align__(16) bf16 sB[64 * 32];    // [n][k]
  const int t = threadIdx.x;
  const int lane = t & 31, wave = t >> 5;
  const int wm = wave & 3, wn = wave >> 2;      // 4 (M) x 2 (N) waves
  const int cl = lane & 15, half = lane >> 4;
  const int m0 = blockIdx.y * 128;
  const int n0 = blockIdx.x * 64;
  v8f a00 = {}, a01 = {}, a10 = {}, a11 = {};
  for (int k0 = 0; k0 < Kp; k0 += 32) {
#pragma unroll
    for (int r = 0; r < 2; ++r) {               // A: 128x32 = 512 vec8 copies
      int vid = t + r * 256;
      int m = vid >> 2;
      int kk = (vid & 3) << 3;
      stage_copy_b128(&A[(size_t)(m0 + m) * Kp + k0 + kk], &sA[m * 32 + kk]);
    }
    {                                           // B: 64x32, transpose to [n][k]
      int kk = t >> 3;
      int ng = (t & 7) << 3;
      U128 val = *(const U128*)&Bm[(size_t)(k0 + kk) * GN_COLS + n0 + ng];
      const bf16* pv = (const bf16*)&val;
#pragma unroll
      for (int e = 0; e < 8; ++e) sB[(ng + e) * 32 + kk] = pv[e];
    }
    // Prefetch next K-step tiles (global_prefetch_b8; no counter effects).
    if (k0 + 32 < Kp) {
      __builtin_prefetch(&A[(size_t)(m0 + (t >> 1)) * Kp + k0 + 32], 0, 3);
      __builtin_prefetch(&Bm[(size_t)(k0 + 32 + (t & 31)) * GN_COLS + n0], 0, 3);
    }
    wait_async_lds();
    __syncthreads();
    v16bf af0 = frag_a(sA, wm * 32 + cl, 32, 0);
    v16bf af1 = frag_a(sA, wm * 32 + 16 + cl, 32, 0);
    v16bf bf0 = frag_b(sB, wn * 32 + cl, 32, 0);
    v16bf bf1 = frag_b(sB, wn * 32 + 16 + cl, 32, 0);
    a00 = wmma_bf16(af0, bf0, a00);
    a01 = wmma_bf16(af0, bf1, a01);
    a10 = wmma_bf16(af1, bf0, a10);
    a11 = wmma_bf16(af1, bf1, a11);
    __syncthreads();
  }
  // C/D layout: VGPR v -> M = v + 8*half, N = lane&15 ; store NCHW directly.
#pragma unroll
  for (int vv = 0; vv < 8; ++vv) {
    int co0 = m0 + wm * 32 + vv + 8 * half;
    int co1 = co0 + 16;
    int nA = n0 + wn * 32 + cl;
    int nB = nA + 16;
    int bA = nA >> 6, hwA = nA & 63;
    int bB = nB >> 6, hwB = nB & 63;
    out[((size_t)bA * M + co0) * HW_SZ + hwA] = a00[vv];
    out[((size_t)bB * M + co0) * HW_SZ + hwB] = a01[vv];
    out[((size_t)bA * M + co1) * HW_SZ + hwA] = a10[vv];
    out[((size_t)bB * M + co1) * HW_SZ + hwB] = a11[vv];
  }
}

// ---------------------------------------------------------------------------
// Attention per spatial position hw: L = Q*K^T/sqrt(C); softmax rows; out=P*V.
// One block per hw, 8 waves, WMMA for both GEMMs.
// ---------------------------------------------------------------------------
__global__ void __launch_bounds__(256)
attn_wmma(const float* __restrict__ q, const float* __restrict__ k,
          const float* __restrict__ v, float* __restrict__ vf) {
  const int hw = blockIdx.x;
  const int t = threadIdx.x;
  const int lane = t & 31, wave = t >> 5;
  const int wm = wave & 3, wn = wave >> 2;
  const int cl = lane & 15, half = lane >> 4;
  __shared__ __align__(16) bf16  sQ[64 * 32];   // [i][k]
  __shared__ __align__(16) bf16  sK[64 * 32];   // [j][k]
  __shared__ __align__(16) float sL[64 * 64];   // logits
  __shared__ __align__(16) bf16  sP[64 * 64];   // probs  [i][j]
  __shared__ __align__(16) bf16  sV[64 * 64];   // [c][j] (n-major for frag_b)

  v8f l0 = {}, l1 = {};
  for (int c0 = 0; c0 < CH; c0 += 32) {
#pragma unroll
    for (int r = 0; r < 8; ++r) {
      int e = t * 8 + r;
      int row = e >> 5, kk = e & 31;
      size_t gi = ((size_t)row * CH + c0 + kk) * HW_SZ + hw;
      sQ[e] = (bf16)q[gi];
      sK[e] = (bf16)k[gi];
    }
    __syncthreads();
    v16bf af = frag_a(sQ, wm * 16 + cl, 32, 0);
    v16bf b0 = frag_b(sK, wn * 32 + cl, 32, 0);
    v16bf b1 = frag_b(sK, wn * 32 + 16 + cl, 32, 0);
    l0 = wmma_bf16(af, b0, l0);
    l1 = wmma_bf16(af, b1, l1);
    __syncthreads();
  }
  const float inv_scale = 0.04419417382415922f;  // 1/sqrt(512)
#pragma unroll
  for (int vv = 0; vv < 8; ++vv) {
    int i = wm * 16 + vv + 8 * half;
    sL[i * 64 + wn * 32 + cl]      = l0[vv] * inv_scale;
    sL[i * 64 + wn * 32 + 16 + cl] = l1[vv] * inv_scale;
  }
  __syncthreads();
  if (t < 64) {                                  // row softmax over j
    float mx = -3.4e38f;
    for (int j = 0; j < 64; ++j) mx = fmaxf(mx, sL[t * 64 + j]);
    float sum = 0.f;
    for (int j = 0; j < 64; ++j) { float e = __expf(sL[t * 64 + j] - mx); sL[t * 64 + j] = e; sum += e; }
    float inv = 1.f / sum;
    for (int j = 0; j < 64; ++j) sP[t * 64 + j] = (bf16)(sL[t * 64 + j] * inv);
  }
  __syncthreads();
  for (int c0 = 0; c0 < CH; c0 += 64) {          // out = P[64x64] * V[64x64chunk]
#pragma unroll
    for (int r = 0; r < 16; ++r) {
      int e = t * 16 + r;
      int cc = e >> 6, j = e & 63;
      sV[e] = (bf16)v[((size_t)j * CH + c0 + cc) * HW_SZ + hw];
    }
    __syncthreads();
    v8f o0 = {}, o1 = {};
    v16bf a0  = frag_a(sP, wm * 16 + cl, 64, 0);
    v16bf a1  = frag_a(sP, wm * 16 + cl, 64, 32);
    v16bf b00 = frag_b(sV, wn * 32 + cl, 64, 0);
    v16bf b01 = frag_b(sV, wn * 32 + cl, 64, 32);
    v16bf b10 = frag_b(sV, wn * 32 + 16 + cl, 64, 0);
    v16bf b11 = frag_b(sV, wn * 32 + 16 + cl, 64, 32);
    o0 = wmma_bf16(a0, b00, o0);
    o0 = wmma_bf16(a1, b01, o0);
    o1 = wmma_bf16(a0, b10, o1);
    o1 = wmma_bf16(a1, b11, o1);
#pragma unroll
    for (int vv = 0; vv < 8; ++vv) {
      int i = wm * 16 + vv + 8 * half;
      vf[((size_t)i * CH + c0 + wn * 32 + cl) * HW_SZ + hw]      = o0[vv];
      vf[((size_t)i * CH + c0 + wn * 32 + 16 + cl) * HW_SZ + hw] = o1[vv];
    }
    __syncthreads();
  }
}

// GroupNorm(1, C) over (C,H,W) per sample (biased var), affine, then ReLU.
__global__ void __launch_bounds__(256)
gn_relu(const float* __restrict__ in, const float* __restrict__ gamma,
        const float* __restrict__ beta, float* __restrict__ out) {
  __shared__ float r1[256], r2[256];
  const int b = blockIdx.x, t = threadIdx.x;
  const float* xb = in + (size_t)b * (CH * HW_SZ);
  float a = 0.f, qq = 0.f;
  for (int i = t; i < CH * HW_SZ; i += 256) { float v = xb[i]; a += v; qq += v * v; }
  r1[t] = a; r2[t] = qq;
  __syncthreads();
  for (int o = 128; o > 0; o >>= 1) {
    if (t < o) { r1[t] += r1[t + o]; r2[t] += r2[t + o]; }
    __syncthreads();
  }
  const float inv_n = 1.f / (float)(CH * HW_SZ);
  float mu = r1[0] * inv_n;
  float var = r2[0] * inv_n - mu * mu;
  float rs = rsqrtf(var + 1e-5f);
  for (int i = t; i < CH * HW_SZ; i += 256) {
    int c = i >> 6;
    float v = (xb[i] - mu) * rs * gamma[c] + beta[c];
    out[(size_t)b * (CH * HW_SZ) + i] = v > 0.f ? v : 0.f;
  }
}

// st = concat(status, rois[:,1:5]) broadcast over HW: [64, 6, 8, 8] f32.
__global__ void __launch_bounds__(256)
build_st(const float* __restrict__ status, const float* __restrict__ rois,
         float* __restrict__ st) {
  int idx = blockIdx.x * 256 + threadIdx.x;
  if (idx >= 64 * 6 * 64) return;
  int c = (idx >> 6) % 6;
  int b = idx / (6 * 64);
  st[idx] = (c < 2) ? status[b * 2 + c] : rois[b * 5 + 1 + (c - 2)];
}

__global__ void __launch_bounds__(256)
add_res(const float* __restrict__ a, const float* __restrict__ b,
        float* __restrict__ o) {
  size_t i = (size_t)blockIdx.x * 256 + threadIdx.x;
  o[i] = a[i] + b[i];
}

// ---------------------------------------------------------------------------
extern "C" void kernel_launch(void* const* d_in, const int* in_sizes, int n_in,
                              void* d_out, int out_size, void* d_ws, size_t ws_size,
                              hipStream_t stream) {
  (void)in_sizes; (void)n_in; (void)out_size; (void)ws_size;
  const float* status = (const float*)d_in[0];
  const float* rois   = (const float*)d_in[1];
  const float* bbox   = (const float*)d_in[2];
  const float* W[30];
  for (int i = 3; i < 30; ++i) W[i] = (const float*)d_in[i];
  const float* gamma = W[28];
  const float* beta  = W[29];

  // workspace carve-up
  const size_t KP_MAX = 20736;                  // 2304*9 (already /32)
  char* ws = (char*)d_ws;
  size_t off = 0;
  auto take = [&](size_t bytes) -> char* {
    char* p = ws + off;
    off += (bytes + 255) & ~(size_t)255;
    return p;
  };
  bf16*  Bmat = (bf16*)take(KP_MAX * GN_COLS * sizeof(bf16));
  bf16*  Aw   = (bf16*)take((size_t)CH * KP_MAX * sizeof(bf16));
  float* st   = (float*)take((size_t)64 * 6 * 64 * sizeof(float));
  const size_t TEN = (size_t)64 * CH * HW_SZ * sizeof(float);  // 8 MB
  float* xb   = (float*)take(TEN);
  float* g1   = (float*)take(TEN);
  float* g2   = (float*)take(TEN);
  float* s1   = (float*)take(TEN);
  float* s2   = (float*)take(TEN);
  float* qb   = (float*)take(TEN);
  float* kb   = (float*)take(TEN);
  float* vb   = (float*)take(TEN);
  float* vfb  = (float*)take(TEN);
  float* gnrb = (float*)take(TEN);
  float* cb   = (float*)take(TEN);

  auto conv = [&](const float* wgt, int Cin, const float* xin, float* y,
                  bool do_im2col) {
    int K = Cin * 9;
    int Kp = (K + 31) & ~31;
    if (do_im2col) {
      size_t tot = (size_t)Kp * GN_COLS;
      im2col_bf16<<<dim3((unsigned)((tot + 255) / 256)), 256, 0, stream>>>(
          xin, Bmat, Cin, Kp);
    }
    size_t wt = (size_t)CH * Kp;
    weight_to_bf16<<<dim3((unsigned)((wt + 255) / 256)), 256, 0, stream>>>(
        wgt, Aw, K, Kp);
    gemm_conv_wmma<<<dim3(GN_COLS / 64, CH / 128), 256, 0, stream>>>(
        Aw, Bmat, y, CH, Kp);
  };

  auto block = [&](const float* qin, int qCin, const float* kvin,
                   const float* qw, const float* kw, const float* vw,
                   const float* cw, const float* resid, float* outp) {
    conv(qw, qCin, qin, qb, true);
    conv(kw, CH, kvin, kb, true);
    conv(vw, CH, kvin, vb, false);                 // reuse kv im2col
    attn_wmma<<<HW_SZ, 256, 0, stream>>>(qb, kb, vb, vfb);
    gn_relu<<<64, 256, 0, stream>>>(vfb, gamma, beta, gnrb);
    conv(cw, CH, gnrb, cb, true);
    add_res<<<8192, 256, 0, stream>>>(resid, cb, outp);
  };

  build_st<<<96, 256, 0, stream>>>(status, rois, st);
  conv(W[3], 2304, bbox, xb, true);                // x = reduce(bbox_feat)

  // stage 1 (queries from status)
  block(st, 6, xb, W[4], W[6], W[7], W[10], xb, g1);    // q1,k1,v1,c11
  block(st, 6, xb, W[5], W[8], W[9], W[11], xb, g2);    // q2,k2,v2,c12
  // stage 2 (self queries)
  block(g1, CH, g1, W[12], W[14], W[15], W[18], g1, s1); // q21,k21,v21,c21
  block(g2, CH, g2, W[13], W[16], W[17], W[19], g2, s2); // q22,k22,v22,c22
  // stage 3 (queries from status again) -> d_out
  float* t1 = (float*)d_out;
  float* t2 = (float*)d_out + (size_t)64 * CH * HW_SZ;
  block(st, 6, s1, W[20], W[22], W[23], W[26], s1, t1);  // q31,k31,v31,c31
  block(st, 6, s2, W[21], W[24], W[25], W[27], s2, t2);  // q32,k32,v32,c32
}
